// MultiHeadAttention_13889924235858
// MI455X (gfx1250) — compile-verified, matrix operand order checked by reading, not run on hardware
//
#include <hip/hip_runtime.h>
#include <math.h>

// ---------------- problem constants ----------------
#define NB     8
#define CH     256
#define HEADS  8
#define DHD    256
#define ATTD   (HEADS * DHD)      // 2048
#define LSEQ   1024               // 32*32
#define QKV_O  (3 * ATTD)         // 6144
#define SCALE_F 0.1041666666666667f  // = 5/48

typedef __attribute__((ext_vector_type(16))) __bf16 v16bf;
typedef __attribute__((ext_vector_type(8)))  float  v8f;

// ---------------- WMMA helper ----------------
__device__ __forceinline__ v8f wmma_bf16(v16bf a, v16bf b, v8f c) {
  return __builtin_amdgcn_wmma_f32_16x16x32_bf16(false, a, false, b,
                                                 (short)0, c, false, false);
}

// CDNA5 16-bit A-fragment layout (ISA 7.12.2): lane&15 = row, lane>>4 selects
// K-offset 0/8; per lane the 16 halves cover K = kb..kb+7 and 16+kb..16+kb+7.
// B fragments use the mirrored layout (lane = column), so the same loader
// works when B is stored transposed (N-major, K contiguous). Works for both
// global and LDS sources (addrspace inferred after inlining -> ds_load_b128).
__device__ __forceinline__ v16bf load_frag_b16(const __bf16* __restrict__ base,
                                               int ld, int row0, int k0) {
  int lane = threadIdx.x & 31;
  int r  = lane & 15;
  int kb = (lane >> 4) << 3;
  const __bf16* p = base + (size_t)(row0 + r) * ld + k0 + kb;
  v16bf f;
#pragma unroll
  for (int i = 0; i < 8; ++i) { f[i] = p[i]; f[8 + i] = p[16 + i]; }
  return f;
}

// ---------------- gfx1250 async copy to LDS (ASYNCcnt) ----------------
__device__ __forceinline__ void async_copy_b128(unsigned lds_off, const void* gptr) {
  unsigned long long ga = (unsigned long long)(size_t)gptr;
  asm volatile("global_load_async_to_lds_b128 %0, %1, off"
               :: "v"(lds_off), "v"(ga) : "memory");
}
__device__ __forceinline__ void wait_async0() {
  asm volatile("s_wait_asynccnt 0x0" ::: "memory");
}

// ---------------- kernel P: convert weights f32 -> bf16 once ----------------
__global__ void k_convert_w(const float* __restrict__ w_in, const float* __restrict__ w_out,
                            __bf16* __restrict__ w_in_bf, __bf16* __restrict__ w_out_bf) {
  int idx = blockIdx.x * blockDim.x + threadIdx.x;
  if (idx < QKV_O * CH) w_in_bf[idx] = (__bf16)w_in[idx];
  if (idx < CH * ATTD)  w_out_bf[idx] = (__bf16)w_out[idx];
}

// ---------------- kernel 0: x (n,C,L) f32 -> xT (n,L,C) bf16 ----------------
__global__ void k_transpose_x(const float* __restrict__ x, __bf16* __restrict__ xT) {
  int idx = blockIdx.x * blockDim.x + threadIdx.x;   // over NB*CH*LSEQ
  int l = idx & (LSEQ - 1);
  int c = (idx >> 10) & (CH - 1);
  int n = idx >> 18;
  xT[((size_t)n * LSEQ + l) * CH + c] = (__bf16)x[idx];
}

// ---------------- kernel 1: QKV projection GEMM ----------------
__global__ void k_qkv(const __bf16* __restrict__ w_in_bf, const float* __restrict__ b_in,
                      const __bf16* __restrict__ xT,
                      __bf16* __restrict__ qT, __bf16* __restrict__ kT,
                      __bf16* __restrict__ vv) {
  int lT = blockIdx.x * 16;          // L tile
  int oT = blockIdx.y * 16;          // output-row tile (of 6144)
  int n  = blockIdx.z;
  const __bf16* xb = xT + (size_t)n * LSEQ * CH;

  v8f acc = {};
#pragma unroll
  for (int kk = 0; kk < CH; kk += 32) {
    v16bf a = load_frag_b16(w_in_bf, CH, oT, kk);
    v16bf b = load_frag_b16(xb, CH, lT, kk);
    acc = wmma_bf16(a, b, acc);
  }

  int lane = threadIdx.x & 31;
  int l = lT + (lane & 15);
  int oBase = oT + ((lane >> 4) << 3);   // 8 consecutive rows per lane
  int h   = oBase / (3 * DHD);
  int j   = oBase % (3 * DHD);
  int sec = j >> 8;                      // 0=q 1=k 2=v (DHD==256)
  int d0  = j & (DHD - 1);
  size_t nh = (size_t)n * HEADS + h;

  if (sec == 0) {
    __bf16* dst = qT + (nh * LSEQ + l) * DHD + d0;
#pragma unroll
    for (int r = 0; r < 8; ++r)
      dst[r] = (__bf16)((acc[r] + b_in[oBase + r]) * SCALE_F);
  } else if (sec == 1) {
    __bf16* dst = kT + (nh * LSEQ + l) * DHD + d0;
#pragma unroll
    for (int r = 0; r < 8; ++r)
      dst[r] = (__bf16)((acc[r] + b_in[oBase + r]) * SCALE_F);
  } else {
    __bf16* dst = vv + (nh * DHD + d0) * LSEQ + l;
#pragma unroll
    for (int r = 0; r < 8; ++r)
      dst[(size_t)r * LSEQ] = (__bf16)((acc[r] + b_in[oBase + r]) * SCALE_F);
  }
}

// ---------------- kernel 2: per-(nh,l-tile) online max/sum of exp(S) ----------
__global__ void k_att_stats(const __bf16* __restrict__ qT, const __bf16* __restrict__ kT,
                            float2* __restrict__ partial) {
  int lT = blockIdx.x * 16;
  int nh = blockIdx.y;
  const __bf16* qb = qT + (size_t)nh * LSEQ * DHD;
  const __bf16* kb = kT + (size_t)nh * LSEQ * DHD;

  v16bf aq[8];
#pragma unroll
  for (int kk = 0; kk < 8; ++kk) aq[kk] = load_frag_b16(qb, DHD, lT, kk * 32);

  float m = -INFINITY, s = 0.0f;
  for (int mt = 0; mt < LSEQ / 16; ++mt) {
    v8f acc = {};
#pragma unroll
    for (int kk = 0; kk < 8; ++kk) {
      v16bf b = load_frag_b16(kb, DHD, mt * 16, kk * 32);
      acc = wmma_bf16(aq[kk], b, acc);
    }
    float tmax = acc[0];
#pragma unroll
    for (int r = 1; r < 8; ++r) tmax = fmaxf(tmax, acc[r]);
    float nm = fmaxf(m, tmax);
    s *= __expf(m - nm);
#pragma unroll
    for (int r = 0; r < 8; ++r) s += __expf(acc[r] - nm);
    m = nm;
  }
#pragma unroll
  for (int off = 16; off > 0; off >>= 1) {
    float om = __shfl_xor(m, off, 32);
    float os = __shfl_xor(s, off, 32);
    float nm = fmaxf(m, om);
    s = s * __expf(m - nm) + os * __expf(om - nm);
    m = nm;
  }
  if ((threadIdx.x & 31) == 0) partial[(size_t)nh * (LSEQ / 16) + blockIdx.x] = make_float2(m, s);
}

// ---------------- kernel 3: reduce 64 partials -> (M, Z) per nh --------------
__global__ void k_stats_reduce(const float2* __restrict__ partial, float2* __restrict__ stats) {
  int nh = blockIdx.x;
  int lane = threadIdx.x;       // 32 threads
  float2 p0 = partial[(size_t)nh * 64 + lane];
  float2 p1 = partial[(size_t)nh * 64 + 32 + lane];
  float m = fmaxf(p0.x, p1.x);
  float s = p0.y * __expf(p0.x - m) + p1.y * __expf(p1.x - m);
#pragma unroll
  for (int off = 16; off > 0; off >>= 1) {
    float om = __shfl_xor(m, off, 32);
    float os = __shfl_xor(s, off, 32);
    float nm = fmaxf(m, om);
    s = s * __expf(m - nm) + os * __expf(om - nm);
    m = nm;
  }
  if (lane == 0) stats[nh] = make_float2(m, s);
}

// ---------------- kernel 4: fused  out = (1/Z) * V * exp(S - M)  -------------
// 4 waves per block, one m-tile each, same (n,h). The q-block (32x256 bf16)
// and v-block (256x32 bf16) for each l-step are shared by all 4 waves, so they
// are staged in LDS with gfx1250 async copies (double buffered; copy of step
// t+1 overlaps the 32 WMMAs of step t). S-tile D-fragments map 1:1 onto the
// B-fragment layout of the V*P WMMA -> exp + cvt in registers, no shuffles.
__global__ void __launch_bounds__(128)
k_att_out(const __bf16* __restrict__ qT, const __bf16* __restrict__ kT,
          const __bf16* __restrict__ vv, const float2* __restrict__ stats,
          __bf16* __restrict__ outT) {
  __shared__ __bf16 sq[2][32 * DHD];   // 2 x 16 KB: q rows l0..l0+31, ld=256
  __shared__ __bf16 sv[2][DHD * 32];   // 2 x 16 KB: v rows d=0..255,  ld=32

  int tid  = threadIdx.x;              // 0..127
  int wave = tid >> 5;
  int mT = blockIdx.x * 64 + wave * 16;
  int nh = blockIdx.y;
  int n  = nh >> 3, h = nh & 7;
  const __bf16* qb = qT + (size_t)nh * LSEQ * DHD;
  const __bf16* kb = kT + (size_t)nh * LSEQ * DHD;
  const __bf16* vb = vv + (size_t)nh * DHD * LSEQ;

  float2 st = stats[nh];
  float M = st.x, zinv = 1.0f / st.y;

  // preload the k fragments for this wave's m-tile (B of the S GEMM)
  v16bf bk[8];
#pragma unroll
  for (int kk = 0; kk < 8; ++kk) bk[kk] = load_frag_b16(kb, DHD, mT, kk * 32);

  v8f zero = {};
  v8f oacc[16];
#pragma unroll
  for (int dt = 0; dt < 16; ++dt) oacc[dt] = zero;

  // cooperative async stage of q(32x256) + v(256x32) for one l-step
  auto stage = [&](int buf, int l0) {
    unsigned sqo = (unsigned)(size_t)&sq[buf][0];
    unsigned svo = (unsigned)(size_t)&sv[buf][0];
#pragma unroll
    for (int i = 0; i < 8; ++i) {
      int c = tid + i * 128;                 // 1024 chunks of 16B
      int row = c >> 5;                      // 32 chunks per 512B q row
      int cb  = (c & 31) * 16;
      async_copy_b128(sqo + c * 16,
                      (const char*)qb + (size_t)(l0 + row) * (DHD * 2) + cb);
    }
#pragma unroll
    for (int i = 0; i < 8; ++i) {
      int c = tid + i * 128;                 // 1024 chunks of 16B
      int row = c >> 2;                      // 4 chunks per 64B v row
      int cb  = (c & 3) * 16;
      async_copy_b128(svo + c * 16,
                      (const char*)vb + (size_t)row * (LSEQ * 2) + l0 * 2 + cb);
    }
  };

  stage(0, 0);
  for (int lt = 0; lt < LSEQ / 32; ++lt) {
    int cur = lt & 1;
    wait_async0();
    __syncthreads();                         // buf[cur] ready, buf[1-cur] free
    if (lt + 1 < LSEQ / 32) stage(1 - cur, (lt + 1) * 32);

    v8f s1 = zero, s2 = zero;
#pragma unroll
    for (int kk = 0; kk < 8; ++kk) {
      v16bf aq1 = load_frag_b16(&sq[cur][0], DHD, 0, kk * 32);
      s1 = wmma_bf16(aq1, bk[kk], s1);
    }
#pragma unroll
    for (int kk = 0; kk < 8; ++kk) {
      v16bf aq2 = load_frag_b16(&sq[cur][0], DHD, 16, kk * 32);
      s2 = wmma_bf16(aq2, bk[kk], s2);
    }
    // exp in registers; D layout of s1/s2 == B fragment layout for V*P
    v16bf bp;
#pragma unroll
    for (int i = 0; i < 8; ++i) {
      bp[i]     = (__bf16)__expf(s1[i] - M);
      bp[8 + i] = (__bf16)__expf(s2[i] - M);
    }
#pragma unroll
    for (int dt = 0; dt < 16; ++dt) {
      v16bf av = load_frag_b16(&sv[cur][0], 32, dt * 16, 0);   // A = V, K=l
      oacc[dt] = wmma_bf16(av, bp, oacc[dt]);
    }
  }

  // write outT (n, L, ATT) bf16, scaled by 1/Z; contiguous 8 elems per lane
  int lane = tid & 31;
  int mcol = mT + (lane & 15);
  int dOff = (lane >> 4) << 3;
#pragma unroll
  for (int dt = 0; dt < 16; ++dt) {
    __bf16* dst = outT + ((size_t)n * LSEQ + mcol) * ATTD + h * DHD + dt * 16 + dOff;
#pragma unroll
    for (int r = 0; r < 8; ++r) dst[r] = (__bf16)(oacc[dt][r] * zinv);
  }
}

// ---------------- kernel 5: output projection ----------------
__global__ void k_proj(const __bf16* __restrict__ w_out_bf, const float* __restrict__ b_out,
                       const __bf16* __restrict__ outT, float* __restrict__ out) {
  int lT = blockIdx.x * 16;
  int cT = blockIdx.y * 16;
  int n  = blockIdx.z;
  const __bf16* ob = outT + (size_t)n * LSEQ * ATTD;

  v8f acc = {};
  for (int kk = 0; kk < ATTD; kk += 32) {
    v16bf a = load_frag_b16(w_out_bf, ATTD, cT, kk);
    v16bf b = load_frag_b16(ob, ATTD, lT, kk);
    acc = wmma_bf16(a, b, acc);
  }

  int lane = threadIdx.x & 31;
  int l = lT + (lane & 15);
  int c0 = cT + ((lane >> 4) << 3);
#pragma unroll
  for (int r = 0; r < 8; ++r)
    out[((size_t)n * CH + c0 + r) * LSEQ + l] = acc[r] + b_out[c0 + r];
}

// ---------------- host side ----------------
extern "C" void kernel_launch(void* const* d_in, const int* in_sizes, int n_in,
                              void* d_out, int out_size, void* d_ws, size_t ws_size,
                              hipStream_t stream) {
  (void)in_sizes; (void)n_in; (void)out_size; (void)ws_size;
  const float* x     = (const float*)d_in[0];
  const float* w_in  = (const float*)d_in[1];
  const float* b_in  = (const float*)d_in[2];
  const float* w_out = (const float*)d_in[3];
  const float* b_out = (const float*)d_in[4];
  float* out = (float*)d_out;

  char* p = (char*)d_ws;
  __bf16* xT      = (__bf16*)p; p += (size_t)NB * LSEQ * CH * 2;            //   4 MB
  __bf16* qT      = (__bf16*)p; p += (size_t)NB * HEADS * LSEQ * DHD * 2;   //  32 MB
  __bf16* kT      = (__bf16*)p; p += (size_t)NB * HEADS * LSEQ * DHD * 2;   //  32 MB
  __bf16* vv      = (__bf16*)p; p += (size_t)NB * HEADS * LSEQ * DHD * 2;   //  32 MB
  __bf16* outT    = (__bf16*)p; p += (size_t)NB * LSEQ * ATTD * 2;          //  32 MB
  __bf16* w_in_bf = (__bf16*)p; p += (size_t)QKV_O * CH * 2;                //   3 MB
  __bf16* w_out_bf= (__bf16*)p; p += (size_t)CH * ATTD * 2;                 //   1 MB
  float2* partial = (float2*)p; p += (size_t)64 * 64 * sizeof(float2);
  float2* stats   = (float2*)p;

  // P) weights f32 -> bf16 (once)
  k_convert_w<<<(QKV_O * CH) / 256, 256, 0, stream>>>(w_in, w_out, w_in_bf, w_out_bf);

  // 0) x -> xT bf16
  k_transpose_x<<<(NB * CH * LSEQ) / 256, 256, 0, stream>>>(x, xT);

  // 1) QKV GEMM (6144 x 1024 per batch, K=256)
  k_qkv<<<dim3(LSEQ / 16, QKV_O / 16, NB), 32, 0, stream>>>(w_in_bf, b_in, xT, qT, kT, vv);

  // 2) global-softmax statistics (online max/sum per (nh, l-tile))
  k_att_stats<<<dim3(LSEQ / 16, NB * HEADS), 32, 0, stream>>>(qT, kT, partial);

  // 3) reduce partials -> (M, Z) per (n,h)
  k_stats_reduce<<<NB * HEADS, 32, 0, stream>>>(partial, stats);

  // 4) fused S -> exp -> V*P with double-buffered async LDS staging
  k_att_out<<<dim3(LSEQ / 64, NB * HEADS), 128, 0, stream>>>(qT, kT, vv, stats, outT);

  // 5) output projection (256 x 1024 per batch, K=2048) + bias
  k_proj<<<dim3(LSEQ / 16, CH / 16, NB), 32, 0, stream>>>(w_out_bf, b_out, outT, out);
}
